// MoE_29738353558256
// MI455X (gfx1250) — matrix operand order repaired and checked by reading, not compiled, not run on hardware
//
#include <hip/hip_runtime.h>

// ---------------- problem constants ----------------
constexpr int NTOK = 4096;   // tokens
constexpr int DIM  = 1024;   // input dim
constexpr int HID  = 4096;   // hidden dim
constexpr int ODIM = 1024;   // output dim
constexpr int NE   = 8;      // experts
constexpr int TOPK = 2;

constexpr int LDT = 40;      // padded LDS row stride (bf16 elems): 80B -> conflict-free

typedef __bf16 bf16_t;
typedef __attribute__((ext_vector_type(16))) __bf16      v16bf;
typedef __attribute__((ext_vector_type(8)))  float       v8f;
typedef __attribute__((ext_vector_type(4)))  unsigned    v4u;
typedef __attribute__((ext_vector_type(4)))  float       v4f;
typedef __attribute__((ext_vector_type(4)))  __bf16      v4bf;
typedef __attribute__((ext_vector_type(2)))  float       v2f;
typedef __attribute__((ext_vector_type(2)))  __bf16      v2bf;

union FragU  { v4u u[2]; v16bf v; };
union Pack4U { v4bf b; unsigned long long q; };
union Pack2U { v2bf b; unsigned u; bf16_t h[2]; };

// float4 -> 4 packed bf16 (two v_cvt_pk_bf16_f32)
static __device__ inline unsigned long long cvt4(float4 f) {
  v4f vf = {f.x, f.y, f.z, f.w};
  Pack4U p;
  p.b = __builtin_convertvector(vf, v4bf);
  return p.q;
}
// float2 -> 2 packed bf16 (one v_cvt_pk_bf16_f32)
static __device__ inline Pack2U cvt2(float a, float b) {
  v2f vf = {a, b};
  Pack2U p;
  p.b = __builtin_convertvector(vf, v2bf);
  return p;
}

// ---------------- init: zero out + counts ----------------
__global__ void moe_zero(float* __restrict__ out, int* __restrict__ counts) {
  size_t i = (size_t)blockIdx.x * blockDim.x + threadIdx.x;  // over N*O/4
  float4 z = make_float4(0.f, 0.f, 0.f, 0.f);
  ((float4*)out)[i] = z;
  if (blockIdx.x == 0 && threadIdx.x < NE) counts[threadIdx.x] = 0;
}

// ---------------- gate: relu -> softmax -> top2 ----------------
__global__ __launch_bounds__(256)
void moe_gate(const float* __restrict__ x, const float* __restrict__ Wg,
              int* __restrict__ counts, int* __restrict__ sel, float* __restrict__ selw) {
  const int wave = threadIdx.x >> 5;
  const int lane = threadIdx.x & 31;
  const int n = blockIdx.x * 8 + wave;
  if (n >= NTOK) return;
  const float* xr = x + (size_t)n * DIM;

  float acc[NE];
#pragma unroll
  for (int e = 0; e < NE; ++e) acc[e] = 0.f;

  for (int i = 0; i < DIM / 32; ++i) {
    const int d = i * 32 + lane;
    const float xv = xr[d];
    const float4* wr4 = (const float4*)(Wg + (size_t)d * NE);
    const float4 w0 = wr4[0], w1 = wr4[1];
    acc[0] += xv * w0.x; acc[1] += xv * w0.y; acc[2] += xv * w0.z; acc[3] += xv * w0.w;
    acc[4] += xv * w1.x; acc[5] += xv * w1.y; acc[6] += xv * w1.z; acc[7] += xv * w1.w;
  }
#pragma unroll
  for (int off = 16; off > 0; off >>= 1)
#pragma unroll
    for (int e = 0; e < NE; ++e) acc[e] += __shfl_xor(acc[e], off, 32);

  if (lane == 0) {
    float mx = 0.f;
#pragma unroll
    for (int e = 0; e < NE; ++e) { acc[e] = fmaxf(acc[e], 0.f); mx = fmaxf(mx, acc[e]); }
    float p[NE], s = 0.f;
#pragma unroll
    for (int e = 0; e < NE; ++e) { p[e] = __expf(acc[e] - mx); s += p[e]; }
    const float inv = 1.f / s;
#pragma unroll
    for (int e = 0; e < NE; ++e) p[e] *= inv;

    int i0 = 0; float p0 = p[0];
#pragma unroll
    for (int e = 1; e < NE; ++e) if (p[e] > p0) { p0 = p[e]; i0 = e; }
    int i1 = -1; float p1 = -1.f;
#pragma unroll
    for (int e = 0; e < NE; ++e) if (e != i0 && p[e] > p1) { p1 = p[e]; i1 = e; }

    const float wsum = p0 + p1;
    sel[2 * n + 0] = i0;            selw[2 * n + 0] = p0 / wsum;
    sel[2 * n + 1] = i1;            selw[2 * n + 1] = p1 / wsum;
    atomicAdd(&counts[i0], 1);
    atomicAdd(&counts[i1], 1);
  }
}

// ---------------- offsets: prefix sum over 8 experts ----------------
__global__ void moe_offsets(const int* __restrict__ counts,
                            int* __restrict__ offsets, int* __restrict__ cursors) {
  if (threadIdx.x == 0 && blockIdx.x == 0) {
    int o = 0;
    for (int e = 0; e < NE; ++e) { offsets[e] = o; o += counts[e]; cursors[e] = 0; }
  }
}

// ---------------- scatter: compact routed rows ----------------
__global__ void moe_scatter(const int* __restrict__ sel, const float* __restrict__ selw,
                            const int* __restrict__ offsets, int* __restrict__ cursors,
                            int* __restrict__ row_token, float* __restrict__ row_w) {
  const int n = blockIdx.x * blockDim.x + threadIdx.x;
  if (n >= NTOK) return;
#pragma unroll
  for (int k = 0; k < TOPK; ++k) {
    const int e = sel[2 * n + k];
    const int pos = atomicAdd(&cursors[e], 1);
    const int r = offsets[e] + pos;
    row_token[r] = n;
    row_w[r] = selw[2 * n + k];
  }
}

// ---------------- pass 1: h = relu(x @ W1[e] + b1[e]) (bf16 out) ----------------
__global__ __launch_bounds__(256)
void moe_expert1(const float* __restrict__ x, const float* __restrict__ W1,
                 const float* __restrict__ b1,
                 const int* __restrict__ counts, const int* __restrict__ offsets,
                 const int* __restrict__ row_token, bf16_t* __restrict__ hbuf) {
  const int e = blockIdx.z;
  const int cnt = counts[e];
  const int mbase = blockIdx.y * 128;
  if (mbase >= cnt) return;
  const int off = offsets[e];
  const int n0 = blockIdx.x * 128;
  const int rows = min(128, cnt - mbase);
  const int t = threadIdx.x;

  __shared__ bf16_t ldsA[128 * LDT];
  __shared__ bf16_t ldsB[128 * LDT];

  // A staging: 4 iters, each thread covers row=(t>>3)+32*it, cols c4..c4+3
  int tokA[4];
#pragma unroll
  for (int it = 0; it < 4; ++it) {
    const int row = (t >> 3) + 32 * it;
    tokA[it] = (row < rows) ? row_token[off + mbase + row] : -1;
  }
  const int c4 = (t & 7) * 4;
  // B staging: 4 iters, k=(t>>5)+8*it, cols bn4..bn4+3 (transposed into LDS [n][k])
  const int bk = t >> 5;
  const int bn4 = (t & 31) * 4;

  const size_t wbase = (size_t)e * DIM * HID;

  v8f acc[4][2];
#pragma unroll
  for (int mi = 0; mi < 4; ++mi)
#pragma unroll
    for (int ni = 0; ni < 2; ++ni)
#pragma unroll
      for (int r = 0; r < 8; ++r) acc[mi][ni][r] = 0.f;

  const int wave = t >> 5, lane = t & 31;
  const int wr = wave >> 2, wc = wave & 3;
  const int lm = lane & 15, khalf = lane >> 4;

  float4 bufA[4], bufB[4];
  auto loadA = [&](int k0) {
#pragma unroll
    for (int it = 0; it < 4; ++it) {
      bufA[it] = make_float4(0.f, 0.f, 0.f, 0.f);
      if (tokA[it] >= 0)
        bufA[it] = *(const float4*)(x + (size_t)tokA[it] * DIM + k0 + c4);
    }
  };
  auto loadB = [&](int k0) {
#pragma unroll
    for (int it = 0; it < 4; ++it) {
      const int k = bk + 8 * it;
      bufB[it] = *(const float4*)(W1 + wbase + (size_t)(k0 + k) * HID + n0 + bn4);
    }
  };

  loadA(0);
  loadB(0);

  for (int k0 = 0; k0 < DIM; k0 += 32) {
    __syncthreads();
    // regs -> LDS (packed cvt: 2x v_cvt_pk_bf16_f32 per float4)
#pragma unroll
    for (int it = 0; it < 4; ++it) {
      const int row = (t >> 3) + 32 * it;
      *(unsigned long long*)(&ldsA[row * LDT + c4]) = cvt4(bufA[it]);
    }
#pragma unroll
    for (int it = 0; it < 4; ++it) {
      const int k = bk + 8 * it;
      const Pack2U p01 = cvt2(bufB[it].x, bufB[it].y);   // ds_store_b16 + _d16_hi
      const Pack2U p23 = cvt2(bufB[it].z, bufB[it].w);
      ldsB[(bn4 + 0) * LDT + k] = p01.h[0];
      ldsB[(bn4 + 1) * LDT + k] = p01.h[1];
      ldsB[(bn4 + 2) * LDT + k] = p23.h[0];
      ldsB[(bn4 + 3) * LDT + k] = p23.h[1];
    }
    __syncthreads();

    // issue next tile's global loads; latency hidden behind WMMA below
    if (k0 + 32 < DIM) { loadA(k0 + 32); loadB(k0 + 32); }

    v16bf a[4], b[2];
#pragma unroll
    for (int mi = 0; mi < 4; ++mi) {      // A 16x32 lane layout
      const bf16_t* p = &ldsA[(wr * 64 + mi * 16 + lm) * LDT];
      FragU f;
      f.u[0] = *(const v4u*)(p + 8 * khalf);
      f.u[1] = *(const v4u*)(p + 16 + 8 * khalf);
      a[mi] = f.v;
    }
#pragma unroll
    for (int ni = 0; ni < 2; ++ni) {      // B 32x16 lane layout
      const bf16_t* p = &ldsB[(wc * 32 + ni * 16 + lm) * LDT];
      FragU f;
      f.u[0] = *(const v4u*)(p + 8 * khalf);
      f.u[1] = *(const v4u*)(p + 16 + 8 * khalf);
      b[ni] = f.v;
    }
#pragma unroll
    for (int mi = 0; mi < 4; ++mi)
#pragma unroll
      for (int ni = 0; ni < 2; ++ni)
        acc[mi][ni] = __builtin_amdgcn_wmma_f32_16x16x32_bf16(
            false, a[mi], false, b[ni], (short)0, acc[mi][ni], false, false);
  }

  // epilogue: bias + relu, store bf16 h
#pragma unroll
  for (int ni = 0; ni < 2; ++ni) {
    const int hcol = n0 + wc * 32 + ni * 16 + lm;
    const float bias = b1[(size_t)e * HID + hcol];
#pragma unroll
    for (int mi = 0; mi < 4; ++mi) {
#pragma unroll
      for (int r = 0; r < 8; ++r) {
        const int m = mbase + wr * 64 + mi * 16 + khalf * 8 + r;
        if (m < cnt) {
          const float val = fmaxf(acc[mi][ni][r] + bias, 0.f);
          hbuf[(size_t)(off + m) * HID + hcol] = (bf16_t)val;
        }
      }
    }
  }
}

// ---------------- pass 2: out[tok] += w * relu(h @ W2[e] + b2[e]) ----------------
__global__ __launch_bounds__(256)
void moe_expert2(const bf16_t* __restrict__ hbuf, const float* __restrict__ W2,
                 const float* __restrict__ b2,
                 const int* __restrict__ counts, const int* __restrict__ offsets,
                 const int* __restrict__ row_token, const float* __restrict__ row_w,
                 float* __restrict__ out) {
  const int e = blockIdx.z;
  const int cnt = counts[e];
  const int mbase = blockIdx.y * 128;
  if (mbase >= cnt) return;
  const int off = offsets[e];
  const int n0 = blockIdx.x * 128;
  const int rows = min(128, cnt - mbase);
  const int t = threadIdx.x;

  __shared__ bf16_t ldsA[128 * LDT];
  __shared__ bf16_t ldsB[128 * LDT];

  const int ar0 = t >> 2;            // A staging: 2 iters, row=ar0+64*it, 16B chunk
  const int achunk = (t & 3) * 8;
  const int bk = t >> 5;
  const int bn4 = (t & 31) * 4;
  const size_t wbase = (size_t)e * HID * ODIM;

  v8f acc[4][2];
#pragma unroll
  for (int mi = 0; mi < 4; ++mi)
#pragma unroll
    for (int ni = 0; ni < 2; ++ni)
#pragma unroll
      for (int r = 0; r < 8; ++r) acc[mi][ni][r] = 0.f;

  const int wave = t >> 5, lane = t & 31;
  const int wr = wave >> 2, wc = wave & 3;
  const int lm = lane & 15, khalf = lane >> 4;

  v4u  bufA[2];
  float4 bufB[4];
  auto loadA = [&](int k0) {
#pragma unroll
    for (int it = 0; it < 2; ++it) {
      const int row = ar0 + 64 * it;
      bufA[it] = (v4u){0u, 0u, 0u, 0u};
      if (row < rows)
        bufA[it] = *(const v4u*)(hbuf + (size_t)(off + mbase + row) * HID + k0 + achunk);
    }
  };
  auto loadB = [&](int k0) {
#pragma unroll
    for (int it = 0; it < 4; ++it) {
      const int k = bk + 8 * it;
      bufB[it] = *(const float4*)(W2 + wbase + (size_t)(k0 + k) * ODIM + n0 + bn4);
    }
  };

  loadA(0);
  loadB(0);

  for (int k0 = 0; k0 < HID; k0 += 32) {
    __syncthreads();
#pragma unroll
    for (int it = 0; it < 2; ++it) {
      const int row = ar0 + 64 * it;
      *(v4u*)(&ldsA[row * LDT + achunk]) = bufA[it];
    }
#pragma unroll
    for (int it = 0; it < 4; ++it) {
      const int k = bk + 8 * it;
      const Pack2U p01 = cvt2(bufB[it].x, bufB[it].y);
      const Pack2U p23 = cvt2(bufB[it].z, bufB[it].w);
      ldsB[(bn4 + 0) * LDT + k] = p01.h[0];
      ldsB[(bn4 + 1) * LDT + k] = p01.h[1];
      ldsB[(bn4 + 2) * LDT + k] = p23.h[0];
      ldsB[(bn4 + 3) * LDT + k] = p23.h[1];
    }
    __syncthreads();

    if (k0 + 32 < HID) { loadA(k0 + 32); loadB(k0 + 32); }

    v16bf a[4], b[2];
#pragma unroll
    for (int mi = 0; mi < 4; ++mi) {
      const bf16_t* p = &ldsA[(wr * 64 + mi * 16 + lm) * LDT];
      FragU f;
      f.u[0] = *(const v4u*)(p + 8 * khalf);
      f.u[1] = *(const v4u*)(p + 16 + 8 * khalf);
      a[mi] = f.v;
    }
#pragma unroll
    for (int ni = 0; ni < 2; ++ni) {
      const bf16_t* p = &ldsB[(wc * 32 + ni * 16 + lm) * LDT];
      FragU f;
      f.u[0] = *(const v4u*)(p + 8 * khalf);
      f.u[1] = *(const v4u*)(p + 16 + 8 * khalf);
      b[ni] = f.v;
    }
#pragma unroll
    for (int mi = 0; mi < 4; ++mi)
#pragma unroll
      for (int ni = 0; ni < 2; ++ni)
        acc[mi][ni] = __builtin_amdgcn_wmma_f32_16x16x32_bf16(
            false, a[mi], false, b[ni], (short)0, acc[mi][ni], false, false);
  }

#pragma unroll
  for (int ni = 0; ni < 2; ++ni) {
    const int ocol = n0 + wc * 32 + ni * 16 + lm;
    const float bias = b2[(size_t)e * ODIM + ocol];
#pragma unroll
    for (int mi = 0; mi < 4; ++mi) {
#pragma unroll
      for (int r = 0; r < 8; ++r) {
        const int m = mbase + wr * 64 + mi * 16 + khalf * 8 + r;
        if (m < cnt) {
          const int tok = row_token[off + m];
          const float w = row_w[off + m];
          const float val = fmaxf(acc[mi][ni][r] + bias, 0.f) * w;
          atomicAdd(out + (size_t)tok * ODIM + ocol, val);
        }
      }
    }
  }
}

// ---------------- launch ----------------
extern "C" void kernel_launch(void* const* d_in, const int* in_sizes, int n_in,
                              void* d_out, int out_size, void* d_ws, size_t ws_size,
                              hipStream_t stream) {
  const float* x  = (const float*)d_in[0];
  const float* Wg = (const float*)d_in[1];
  const float* W1 = (const float*)d_in[2];
  const float* b1 = (const float*)d_in[3];
  const float* W2 = (const float*)d_in[4];
  const float* b2 = (const float*)d_in[5];
  float* out = (float*)d_out;

  char* ws = (char*)d_ws;
  int*    counts    = (int*)(ws + 0);
  int*    cursors   = (int*)(ws + 64);
  int*    offsets   = (int*)(ws + 128);
  int*    sel       = (int*)(ws + 256);
  float*  selw      = (float*)(ws + 256 + (size_t)2 * NTOK * 4);
  int*    row_token = (int*)(ws + 256 + (size_t)4 * NTOK * 4);
  float*  row_w     = (float*)(ws + 256 + (size_t)6 * NTOK * 4);
  bf16_t* hbuf      = (bf16_t*)(ws + 256 + (size_t)8 * NTOK * 4);  // 131328B, 256B aligned

  moe_zero<<<dim3((NTOK * ODIM / 4) / 256), 256, 0, stream>>>(out, counts);
  moe_gate<<<dim3(NTOK / 8), 256, 0, stream>>>(x, Wg, counts, sel, selw);
  moe_offsets<<<1, 32, 0, stream>>>(counts, offsets, cursors);
  moe_scatter<<<dim3(NTOK / 256), 256, 0, stream>>>(sel, selw, offsets, cursors, row_token, row_w);
  moe_expert1<<<dim3(HID / 128, NTOK / 128, NE), 256, 0, stream>>>(
      x, W1, b1, counts, offsets, row_token, hbuf);
  moe_expert2<<<dim3(ODIM / 128, NTOK / 128, NE), 256, 0, stream>>>(
      hbuf, W2, b2, counts, offsets, row_token, row_w, out);
}
